// MultiheadAttention_12120397709847
// MI455X (gfx1250) — matrix operand order repaired and checked by reading, not compiled, not run on hardware
//
#include <hip/hip_runtime.h>

// ---------------------------------------------------------------------------
// MultiheadAttention forward for MI455X (gfx1250, wave32, WMMA bf16 path)
// B=4 S=2048 D=1024 H=16 DH=64
// ---------------------------------------------------------------------------

typedef __attribute__((ext_vector_type(16))) __bf16 v16bf;
typedef __attribute__((ext_vector_type(8)))  float  v8f;
typedef __attribute__((vector_size(16)))     int    v4i_t;   // async-copy unit

#define DEVFN static __device__ __forceinline__

constexpr int CB  = 4;     // batch
constexpr int CS  = 2048;  // sequence
constexpr int CD  = 1024;  // model dim
constexpr int CH  = 16;    // heads
constexpr int CDH = 64;    // head dim

// ---- gfx1250 async global->LDS copy (ASYNCcnt-tracked), guarded -----------
#if defined(__has_builtin)
#  if __has_builtin(__builtin_amdgcn_global_load_async_to_lds_b128)
#    define HAVE_ASYNC_LDS 1
#  endif
#endif
#ifndef HAVE_ASYNC_LDS
#  define HAVE_ASYNC_LDS 0
#endif

DEVFN void async_copy_b128(const __bf16* gsrc, __bf16* ldst) {
#if HAVE_ASYNC_LDS
    // per-lane 16B copy, global -> LDS, no VGPR round trip
    __builtin_amdgcn_global_load_async_to_lds_b128(
        (v4i_t*)gsrc, (v4i_t*)ldst, 0, 0);
#else
    *(uint4*)ldst = *(const uint4*)gsrc;
#endif
}

DEVFN void wait_async() {
#if HAVE_ASYNC_LDS
#  if __has_builtin(__builtin_amdgcn_s_wait_asynccnt)
    __builtin_amdgcn_s_wait_asynccnt(0);
#  else
    asm volatile("s_wait_asynccnt 0x0" ::: "memory");
#  endif
#endif
}

// f32 -> bf16, round-to-nearest-even (bit math: no reliance on cvt ISA)
DEVFN __bf16 tobf(float f) {
    unsigned u = __builtin_bit_cast(unsigned, f);
    unsigned r = u + 0x7FFFu + ((u >> 16) & 1u);
    return __builtin_bit_cast(__bf16, (unsigned short)(r >> 16));
}

DEVFN void put_pair(v16bf& v, int j2, unsigned u) {
    v[2 * j2]     = __builtin_bit_cast(__bf16, (unsigned short)(u & 0xFFFFu));
    v[2 * j2 + 1] = __builtin_bit_cast(__bf16, (unsigned short)(u >> 16));
}

// 16-bit A-matrix 16x32 fragment (ISA 7.12.2):
//  lanes 0-15 : M=lane,    elem j -> K = j + (j>=8 ? 8 : 0)
//  lanes 16-31: M=lane-16, elem j -> K = j + 8 + (j>=8 ? 8 : 0)
// source is row-major [row][k] with `pitch` elements per row.
DEVFN v16bf load_frag_a(const __bf16* base, int pitch, int lane, int row_off, int k_off) {
    const __bf16* rp = base + (size_t)((lane & 15) + row_off) * pitch
                            + k_off + ((lane & 16) ? 8 : 0);
    v16bf v;
#pragma unroll
    for (int j2 = 0; j2 < 8; ++j2) {
        const int k2 = ((j2 & 4) << 2) + ((j2 & 3) << 1);   // 0,2,4,6,16,18,20,22
        put_pair(v, j2, *(const unsigned*)(rp + k2));
    }
    return v;
}

// 16-bit B-matrix 32x16 fragment (same K order as A fragment):
//  lanes 0-15 : N=lane,    elem j -> K = j + (j>=8 ? 8 : 0)
//  lanes 16-31: N=lane-16, elem j -> K = j + 8 + (j>=8 ? 8 : 0)
// source is [n][k] row-major with `pitch` elements per row.
DEVFN v16bf load_frag_b(const __bf16* base, int pitch, int lane, int row_off, int k_off) {
    const __bf16* rp = base + (size_t)((lane & 15) + row_off) * pitch
                            + k_off + ((lane & 16) ? 8 : 0);
    v16bf v;
#pragma unroll
    for (int j2 = 0; j2 < 8; ++j2)
        put_pair(v, j2, *(const unsigned*)(rp + ((j2 & 4) << 2) + ((j2 & 3) << 1)));
    return v;
}

DEVFN v8f wmma_bf16(v16bf a, v16bf b, v8f c) {
    return __builtin_amdgcn_wmma_f32_16x16x32_bf16(
        /*neg_a=*/false, a, /*neg_b=*/false, b,
        /*c_mod=*/(short)0, c, /*reuse_a=*/false, /*reuse_b=*/false);
}

// ---------------------------------------------------------------------------
// GEMM: C[M,N] = scale * (A[M,K] @ W[K,N])
//   A: f32 global (converted to bf16) or bf16 workspace (async-copied)
//   W: f32 global (converted to bf16, transposed into LDS)
//   C: bf16 workspace or f32 out
// block = 256 threads (8 waves); tile 128(M) x 128(N); K staged 32-wide,
// double-buffered in LDS (one barrier per K-step, stage k+1 overlaps wmma k).
// waves: 4 along M x 2 along N, each wave 32x64 = 2x4 WMMA frags
// ---------------------------------------------------------------------------
template <bool A_BF16, bool OUT_BF16>
__global__ __launch_bounds__(256) void gemm_bf16_wmma(
    const void* __restrict__ Aptr, const float* __restrict__ W,
    void* __restrict__ Cptr, int M, int N, int K, float scale)
{
    constexpr int AP = 40;                    // LDS pitch (bf16), 80B = 5x16B
    __shared__ __bf16 a_lds[2][128 * AP];     // [m][k]
    __shared__ __bf16 w_lds[2][128 * AP];     // [n][k]  (transposed on store)

    const int tid  = threadIdx.x;
    const int lane = tid & 31;
    const int wave = tid >> 5;
    const int wm   = wave & 3;                // 0..3  -> M offset wm*32
    const int wn   = wave >> 2;               // 0..1  -> N offset wn*64
    const int m0   = blockIdx.y * 128;
    const int n0   = blockIdx.x * 128;

    const float*  Af = (const float*)Aptr;
    const __bf16* Ab = (const __bf16*)Aptr;

    auto stage = [&](int buf, int kb) {
        // ---- A tile: 128 rows x 32 k ----
        if constexpr (A_BF16) {
            // bf16 source: straight async b128 copies (2 x 16B per thread)
#pragma unroll
            for (int i = 0; i < 2; ++i) {
                const int f   = tid * 2 + i;   // 0..511 chunks of 8 bf16
                const int row = f >> 2;        // 0..127
                const int c8  = (f & 3) * 8;   // 0..24
                async_copy_b128(Ab + (size_t)(m0 + row) * K + kb + c8,
                                &a_lds[buf][row * AP + c8]);
            }
        } else {
            // f32 source: convert on the way into LDS
#pragma unroll
            for (int i = 0; i < 4; ++i) {
                const int f   = tid * 4 + i;   // 0..1023 groups of 4 elems
                const int row = f >> 3;        // 0..127
                const int c4  = (f & 7) * 4;   // 0..28
                const float4 v4 = *(const float4*)(Af + (size_t)(m0 + row) * K + kb + c4);
                a_lds[buf][row * AP + c4 + 0] = tobf(v4.x);
                a_lds[buf][row * AP + c4 + 1] = tobf(v4.y);
                a_lds[buf][row * AP + c4 + 2] = tobf(v4.z);
                a_lds[buf][row * AP + c4 + 3] = tobf(v4.w);
            }
        }
        // ---- W tile transposed: [n][k], 32 k-rows x 128 n ----
#pragma unroll
        for (int i = 0; i < 4; ++i) {
            const int f  = tid * 4 + i;        // 0..1023
            const int kr = f >> 5;             // 0..31
            const int n4 = (f & 31) * 4;       // 0..124
            const float4 v4 = *(const float4*)(W + (size_t)(kb + kr) * N + n0 + n4);
            w_lds[buf][(n4 + 0) * AP + kr] = tobf(v4.x);
            w_lds[buf][(n4 + 1) * AP + kr] = tobf(v4.y);
            w_lds[buf][(n4 + 2) * AP + kr] = tobf(v4.z);
            w_lds[buf][(n4 + 3) * AP + kr] = tobf(v4.w);
        }
        // prefetch weight stream two K-tiles ahead -> global_prefetch
        if (kb + 64 < K)
            __builtin_prefetch(W + (size_t)(kb + 64 + (tid & 31)) * N
                                 + n0 + (tid >> 5) * 16, 0, 0);
    };

    v8f acc[2][4];
#pragma unroll
    for (int i = 0; i < 2; ++i)
#pragma unroll
        for (int j = 0; j < 4; ++j) acc[i][j] = v8f{};

    stage(0, 0);

#pragma unroll 1
    for (int kb = 0; kb < K; kb += 32) {
        const int buf = (kb >> 5) & 1;
        wait_async();
        __syncthreads();
        if (kb + 32 < K) stage(buf ^ 1, kb + 32);   // overlaps with wmma below

        v16bf afr[2], bfr[4];
#pragma unroll
        for (int fm = 0; fm < 2; ++fm)
            afr[fm] = load_frag_a(a_lds[buf], AP, lane, wm * 32 + fm * 16, 0);
#pragma unroll
        for (int fn = 0; fn < 4; ++fn)
            bfr[fn] = load_frag_b(w_lds[buf], AP, lane, wn * 64 + fn * 16, 0);
#pragma unroll
        for (int fm = 0; fm < 2; ++fm)
#pragma unroll
            for (int fn = 0; fn < 4; ++fn)
                acc[fm][fn] = wmma_bf16(afr[fm], bfr[fn], acc[fm][fn]);
    }

    // ---- epilogue: C layout -> lane holds col N=lane%16, rows r / r+8 ----
    const int hi = (lane & 16) ? 8 : 0;
#pragma unroll
    for (int fm = 0; fm < 2; ++fm) {
#pragma unroll
        for (int fn = 0; fn < 4; ++fn) {
            const int rbase = m0 + wm * 32 + fm * 16 + hi;
            const int col   = n0 + wn * 64 + fn * 16 + (lane & 15);
#pragma unroll
            for (int r = 0; r < 8; ++r) {
                const float vv = acc[fm][fn][r] * scale;
                if constexpr (OUT_BF16)
                    ((__bf16*)Cptr)[(size_t)(rbase + r) * N + col] = tobf(vv);
                else
                    ((float*)Cptr)[(size_t)(rbase + r) * N + col] = vv;
            }
        }
    }
}

// ---------------------------------------------------------------------------
// Flash attention: per (b, h, 128 q-rows) block; 8 waves x 16 q-rows each.
// Streams 64-wide KV tiles with online softmax; never stores SxS probs.
// q/k/v/ctx live in workspace as bf16 [B*S, D] (head h at cols h*64..h*64+63).
// K tile staged with async b128 global->LDS copies; V transposed in registers.
// ---------------------------------------------------------------------------
__global__ __launch_bounds__(256) void attn_kernel(
    const __bf16* __restrict__ Q, const __bf16* __restrict__ Kw,
    const __bf16* __restrict__ V, const float* __restrict__ bias,
    __bf16* __restrict__ ctx)
{
    constexpr int P = 72;                       // LDS pitch, 144B = 9x16B
    __shared__ __bf16 k_lds[64 * P];            // [kv][dh]   (B-frags for QK^T)
    __shared__ __bf16 v_lds[64 * P];            // [dh][kv]   (B-frags for P@V)
    __shared__ __bf16 p_lds[8 * 16 * P];        // per-wave 16x64 prob strip

    const int tid  = threadIdx.x;
    const int lane = tid & 31;
    const int wave = tid >> 5;
    const int h    = blockIdx.y;
    const int b    = blockIdx.z;
    const int q0   = blockIdx.x * 128 + wave * 16;   // this wave's q rows
    const int hi   = (lane & 16) ? 8 : 0;

    // Q strip 16 x 64 as two A-frags, loaded directly from global (A layout
    // wants contiguous K pairs -> row-major [s][dh] matches exactly)
    v16bf qf[2];
    {
        const __bf16* qbase = Q + (size_t)(b * CS + q0) * CD + h * CDH;
#pragma unroll
        for (int ks = 0; ks < 2; ++ks)
            qf[ks] = load_frag_a(qbase, CD, lane, 0, ks * 32);
    }

    float m_[8], l_[8];
    v8f   o[4];
#pragma unroll
    for (int r = 0; r < 8; ++r) { m_[r] = -__builtin_inff(); l_[r] = 0.f; }
#pragma unroll
    for (int fn = 0; fn < 4; ++fn) o[fn] = v8f{};

#pragma unroll 1
    for (int kv0 = 0; kv0 < CS; kv0 += 64) {
        // ---- K tile [kv][dh]: async b128 global -> LDS (2 chunks/thread) ----
#pragma unroll
        for (int i = 0; i < 2; ++i) {
            const int f   = tid * 2 + i;        // 0..511 chunks of 8 bf16
            const int r   = f >> 3;             // kv row 0..63
            const int c8  = (f & 7) * 8;        // dh col 0..56
            async_copy_b128(Kw + (size_t)(b * CS + kv0 + r) * CD + h * CDH + c8,
                            &k_lds[r * P + c8]);
        }
        // ---- V tile transposed [dh][kv] (register transpose) ----
#pragma unroll
        for (int i = 0; i < 4; ++i) {
            const int f = tid * 4 + i;          // 0..1023 groups of 4 bf16
            const int r = f >> 4;               // kv row 0..63
            const int g = (f & 15) * 4;         // dh col 0..60
            union { uint2 u; __bf16 hh[4]; } cv;
            cv.u = *(const uint2*)(V + (size_t)(b * CS + kv0 + r) * CD + h * CDH + g);
#pragma unroll
            for (int j = 0; j < 4; ++j) v_lds[(g + j) * P + r] = cv.hh[j];
        }
        wait_async();
        __syncthreads();

        // ---- S = Q @ K^T + bias  (16 x 64 strip, 4 C-frags) ----
        v8f sf[4];
#pragma unroll
        for (int fn = 0; fn < 4; ++fn) {
            v8f c = v8f{};
#pragma unroll
            for (int ks = 0; ks < 2; ++ks) {
                v16bf kb = load_frag_b(k_lds, P, lane, fn * 16, ks * 32);
                c = wmma_bf16(qf[ks], kb, c);
            }
            const int col = kv0 + fn * 16 + (lane & 15);
            const float* bp = bias + (size_t)b * CS * CS + (size_t)(q0 + hi) * CS + col;
#pragma unroll
            for (int r = 0; r < 8; ++r) c[r] = c[r] + bp[(size_t)r * CS];
            sf[fn] = c;
        }

        // ---- online softmax: row stats via 16-lane-half shfl_xor ----
        float mn[8];
#pragma unroll
        for (int r = 0; r < 8; ++r) mn[r] = m_[r];
#pragma unroll
        for (int fn = 0; fn < 4; ++fn)
#pragma unroll
            for (int r = 0; r < 8; ++r) mn[r] = fmaxf(mn[r], sf[fn][r]);
#pragma unroll
        for (int off = 1; off < 16; off <<= 1)
#pragma unroll
            for (int r = 0; r < 8; ++r)
                mn[r] = fmaxf(mn[r], __shfl_xor(mn[r], off, 32));

        float al[8], rs[8];
#pragma unroll
        for (int r = 0; r < 8; ++r) {
            al[r] = __expf(m_[r] - mn[r]);
            m_[r] = mn[r];
            rs[r] = 0.f;
        }
#pragma unroll
        for (int fn = 0; fn < 4; ++fn)
#pragma unroll
            for (int r = 0; r < 8; ++r) {
                const float p = __expf(sf[fn][r] - m_[r]);
                rs[r] += p;
                p_lds[(size_t)(wave * 16 + r + hi) * P + fn * 16 + (lane & 15)] = tobf(p);
            }
#pragma unroll
        for (int off = 1; off < 16; off <<= 1)
#pragma unroll
            for (int r = 0; r < 8; ++r) rs[r] += __shfl_xor(rs[r], off, 32);
#pragma unroll
        for (int r = 0; r < 8; ++r) l_[r] = l_[r] * al[r] + rs[r];
#pragma unroll
        for (int fn = 0; fn < 4; ++fn)
#pragma unroll
            for (int r = 0; r < 8; ++r) o[fn][r] = o[fn][r] * al[r];

        // ---- O += P @ V  (per-wave p strip, shared v tile) ----
#pragma unroll
        for (int ks = 0; ks < 2; ++ks) {
            v16bf pa = load_frag_a(p_lds + (size_t)wave * 16 * P, P, lane, 0, ks * 32);
#pragma unroll
            for (int fn = 0; fn < 4; ++fn) {
                v16bf vb = load_frag_b(v_lds, P, lane, fn * 16, ks * 32);
                o[fn] = wmma_bf16(pa, vb, o[fn]);
            }
        }
        __syncthreads();
    }

    // ---- normalize and write context (bf16 workspace) ----
#pragma unroll
    for (int fn = 0; fn < 4; ++fn) {
        const int col = h * CDH + fn * 16 + (lane & 15);
#pragma unroll
        for (int r = 0; r < 8; ++r)
            ctx[(size_t)(b * CS + q0 + r + hi) * CD + col] = tobf(o[fn][r] / l_[r]);
    }
}

// ---------------------------------------------------------------------------
// Launch: q/k/v projections -> flash attention -> output projection
// ---------------------------------------------------------------------------
extern "C" void kernel_launch(void* const* d_in, const int* in_sizes, int n_in,
                              void* d_out, int out_size, void* d_ws, size_t ws_size,
                              hipStream_t stream)
{
    const float* Xq   = (const float*)d_in[0];   // query_antecedent [B,S,D]
    const float* Xm   = (const float*)d_in[1];   // memory_antecedent [B,S,D]
    const float* bias = (const float*)d_in[2];   // [B,S,S]
    const float* Wq   = (const float*)d_in[3];
    const float* Wk   = (const float*)d_in[4];
    const float* Wv   = (const float*)d_in[5];
    const float* Wo   = (const float*)d_in[6];
    float* out = (float*)d_out;

    const int M = CB * CS;   // 8192
    const int N = CD;        // 1024
    const int K = CD;        // 1024
    const size_t bufBytes = (size_t)M * N * sizeof(__bf16);  // 16 MB

    __bf16* qws = (__bf16*)d_ws;
    __bf16* kws = (__bf16*)((char*)d_ws + 1 * bufBytes);
    __bf16* vws = (__bf16*)((char*)d_ws + 2 * bufBytes);
    __bf16* cws = (__bf16*)((char*)d_ws + 3 * bufBytes);

    dim3 blk(256);
    dim3 gg(N / 128, M / 128);
    const float scale = 0.125f;  // DH^-0.5, folded into q projection

    gemm_bf16_wmma<false, true><<<gg, blk, 0, stream>>>(Xq, Wq, qws, M, N, K, scale);
    gemm_bf16_wmma<false, true><<<gg, blk, 0, stream>>>(Xm, Wk, kws, M, N, K, 1.0f);
    gemm_bf16_wmma<false, true><<<gg, blk, 0, stream>>>(Xm, Wv, vws, M, N, K, 1.0f);

    dim3 ga(CS / 128, CH, CB);
    attn_kernel<<<ga, blk, 0, stream>>>(qws, kws, vws, bias, cws);

    gemm_bf16_wmma<true, false><<<gg, blk, 0, stream>>>(cws, Wo, out, M, N, K, 1.0f);
}
